// Copy_56470230008202
// MI455X (gfx1250) — matrix operand order repaired and checked by reading, not compiled, not run on hardware
//
#include <hip/hip_runtime.h>
#include <hip/hip_bf16.h>

// MI455X / gfx1250, wave32. All heavy math via v_wmma_f32_16x16x32_f16.
// GEMMs register-blocked to 32x64 output tiles per wave (2 A-frags x 4
// B-frags -> 8 WMMAs per 32-wide K step) to lift L2 arithmetic intensity.

typedef _Float16 f16;
typedef __attribute__((ext_vector_type(16))) _Float16 v16h;
typedef __attribute__((ext_vector_type(8)))  _Float16 v8h;
typedef __attribute__((ext_vector_type(8)))  float    v8f;

#define H      1024
#define NH     16
#define HD     64
#define CINC   1280
#define VOCABN 32000
#define LIMITN 512
#define LQ     2048
#define SQ     2048
#define PAD    16

__device__ __forceinline__ float selu_f(float x) {
  const float scale = 1.0507009873554805f, alpha = 1.6732632423543772f;
  return x > 0.f ? scale * x : scale * alpha * (__expf(x) - 1.f);
}

__device__ __forceinline__ v8f wmma32(v16h a, v16h b, v8f c) {
  // (neg_a, A, neg_b, B, c_mod, C, reuse_a, reuse_b)
  return __builtin_amdgcn_wmma_f32_16x16x32_f16(false, a, false, b, (short)0, c,
                                                false, false);
}

// A fragment: 16x32 (M=row, K=col), row-major source with given pitch.
// lane m(0-15) holds K = 16*b .. 16*b+15 contiguous, b=lane>>4 (matches load_b).
__device__ __forceinline__ v16h load_a(const f16* tb, int pitch) {
  int lane = threadIdx.x & 31;
  const f16* p = tb + (lane & 15) * pitch + ((lane >> 4) << 4);
  v8h lo = *(const v8h*)p;
  v8h hi = *(const v8h*)(p + 8);
  return __builtin_shufflevector(lo, hi, 0,1,2,3,4,5,6,7,8,9,10,11,12,13,14,15);
}

// B fragment for B^T stored row-major (N rows x K cols): lane n(0-15) holds
// K = 16*(lane>>4) .. +15 contiguous (sparse-B table pattern scaled to K=32).
__device__ __forceinline__ v16h load_b(const f16* tb, int pitch) {
  int lane = threadIdx.x & 31;
  const f16* p = tb + (lane & 15) * pitch + ((lane >> 4) << 4);
  return *(const v16h*)p;
}

// ---------------- weight-norm prep (fold g/||v|| and convert to f16) -------

__global__ __launch_bounds__(256) void wn_rows(const float* v, const float* g,
                                               f16* out, int cols) {
  int row = blockIdx.x;
  __shared__ float red[256];
  float s = 0.f;
  for (int c = threadIdx.x; c < cols; c += 256) {
    float x = v[(long)row * cols + c]; s += x * x;
  }
  red[threadIdx.x] = s; __syncthreads();
  for (int st = 128; st > 0; st >>= 1) {
    if (threadIdx.x < st) red[threadIdx.x] += red[threadIdx.x + st];
    __syncthreads();
  }
  float sc = g[row] * rsqrtf(red[0]);
  for (int c = threadIdx.x; c < cols; c += 256)
    out[(long)row * cols + c] = (f16)(v[(long)row * cols + c] * sc);
}

// Conv1d weight (O, I, 3) -> 3 planes of (O, I) f16, norm over (I,K).
__global__ __launch_bounds__(256) void wn_conv(const float* v, const float* g,
                                               f16* out, int cin, int nOut) {
  int o = blockIdx.x;
  __shared__ float red[256];
  int n = cin * 3;
  float s = 0.f;
  for (int i = threadIdx.x; i < n; i += 256) {
    float x = v[(long)o * n + i]; s += x * x;
  }
  red[threadIdx.x] = s; __syncthreads();
  for (int st = 128; st > 0; st >>= 1) {
    if (threadIdx.x < st) red[threadIdx.x] += red[threadIdx.x + st];
    __syncthreads();
  }
  float sc = g[o] * rsqrtf(red[0]);
  for (int i = threadIdx.x; i < n; i += 256) {
    int c = i / 3, k = i % 3;
    out[(long)k * nOut * cin + (long)o * cin + c] = (f16)(v[(long)o * n + i] * sc);
  }
}

// ---------------- activation prep -----------------------------------------

__global__ __launch_bounds__(256) void prep_o(const float* o, f16* xo) {
  long idx = (long)blockIdx.x * 256 + threadIdx.x;
  long tot = (long)(LQ + 2 * PAD) * CINC;
  if (idx >= tot) return;
  int row = (int)(idx / CINC), c = (int)(idx % CINC);
  int r = row - PAD;
  float v = (r >= 0 && r < LQ) ? o[(long)r * CINC + c] : 0.f;
  xo[idx] = (f16)v;
}

// kf = selu(f) (S,H) f16 ; kt = transposed per-feature (H rows, S cols)
__global__ __launch_bounds__(256) void prep_kv(const float* f, f16* kf, f16* kt) {
  long idx = (long)blockIdx.x * 256 + threadIdx.x;
  if (idx >= (long)SQ * H) return;
  int s = (int)(idx / H), hc = (int)(idx % H);
  float v = selu_f(f[idx]);
  kf[idx] = (f16)v;
  kt[(long)hc * SQ + s] = (f16)v;
}

__global__ __launch_bounds__(256) void zero_h(f16* p, int n) {
  int i = blockIdx.x * 256 + threadIdx.x;
  if (i < n) p[i] = (f16)0.f;
}

// cat[r][H + hp] = q[t*16 + hp/64][n*64 + hp%64], n=r>>7, t=r&127
__global__ __launch_bounds__(256) void q2pack(const f16* q, f16* cat) {
  long idx = (long)blockIdx.x * 256 + threadIdx.x;
  if (idx >= (long)LQ * H) return;
  int r = (int)(idx / H), hp = (int)(idx % H);
  int i = hp >> 6, d = hp & 63;
  int n = r >> 7, t = r & 127;
  cat[(long)r * (2 * H) + H + hp] = q[(long)(t * 16 + i) * H + n * 64 + d];
}

// ---------------- conv as 3 shifted WMMA GEMMs, 32x64 tile per wave -------
// Y(L,nOut) = sum_s Xpad[l-1+s] * W_s^T
__global__ __launch_bounds__(256) void conv_gemm(const f16* Xpad, const f16* W,
                                                 const float* bias, f16* Y,
                                                 int K, int Nblk) {
  int w = (blockIdx.x << 3) + (threadIdx.x >> 5);
  int Mblk = LQ / 32;
  if (w >= Mblk * Nblk) return;
  int tm = w / Nblk, tn = w % Nblk;
  int nOut = Nblk * 64;
  v8f zero = {};
  v8f acc[2][4];
  for (int i = 0; i < 2; ++i)
    for (int n = 0; n < 4; ++n) acc[i][n] = zero;
  for (int k = 0; k < K; k += 32) {
    for (int s = 0; s < 3; ++s) {
      const f16* Ab = Xpad + (long)(tm * 32 + PAD - 1 + s) * K + k;
      v16h a0 = load_a(Ab, K);
      v16h a1 = load_a(Ab + (long)16 * K, K);
      const f16* Bb = W + (long)s * nOut * K + (long)tn * 64 * K + k;
      for (int n = 0; n < 4; ++n) {
        v16h b = load_b(Bb + (long)n * 16 * K, K);
        acc[0][n] = wmma32(a0, b, acc[0][n]);
        acc[1][n] = wmma32(a1, b, acc[1][n]);
      }
    }
  }
  int lane = threadIdx.x & 31;
  int col = lane & 15, rb = (lane >> 4) << 3;
  for (int n = 0; n < 4; ++n) {
    float bi = bias[tn * 64 + n * 16 + col];
    for (int i = 0; i < 2; ++i)
      for (int j = 0; j < 8; ++j) {
        float v = selu_f(acc[i][n][j] + bi);
        Y[(long)(tm * 32 + i * 16 + rb + j) * nOut + tn * 64 + n * 16 + col] =
            (f16)v;
      }
  }
}

// ---------------- generic GEMM: C = A * B^T (+bias)(+selu), 32x64 tile ----
__global__ __launch_bounds__(256) void gemm_ws(const f16* A, int lda,
                                               const f16* B, int ldb,
                                               const float* bias, int K,
                                               int Mblk, int Nblk,
                                               f16* outh, float* outf, int ldc,
                                               int act) {
  int w = (blockIdx.x << 3) + (threadIdx.x >> 5);
  if (w >= Mblk * Nblk) return;
  int tm = w / Nblk, tn = w % Nblk;
  const f16* Ab = A + (long)tm * 32 * lda;
  const f16* Bb = B + (long)tn * 64 * ldb;
  v8f zero = {};
  v8f acc[2][4];
  for (int i = 0; i < 2; ++i)
    for (int n = 0; n < 4; ++n) acc[i][n] = zero;
  for (int k = 0; k < K; k += 32) {
    v16h a0 = load_a(Ab + k, lda);
    v16h a1 = load_a(Ab + (long)16 * lda + k, lda);
    for (int n = 0; n < 4; ++n) {
      v16h b = load_b(Bb + (long)n * 16 * ldb + k, ldb);
      acc[0][n] = wmma32(a0, b, acc[0][n]);
      acc[1][n] = wmma32(a1, b, acc[1][n]);
    }
  }
  int lane = threadIdx.x & 31;
  int col = lane & 15, rb = (lane >> 4) << 3;
  for (int n = 0; n < 4; ++n) {
    float bi = bias[tn * 64 + n * 16 + col];
    for (int i = 0; i < 2; ++i)
      for (int j = 0; j < 8; ++j) {
        float v = acc[i][n][j] + bi;
        if (act) v = selu_f(v);
        long off = (long)(tm * 32 + i * 16 + rb + j) * ldc + tn * 64 + n * 16 + col;
        if (outh) outh[off] = (f16)v;
        else      outf[off] = v;
      }
  }
}

// ---------------- flash attention, one wave per (head, 16-query tile) -----
// Output: cat row r = h*128 + t, cols [i*64 + d] (head-interleaved reshape
// of the reference collapses each wave's 16x64 tile into one contiguous row).
__global__ __launch_bounds__(256) void attn_flash(const f16* Q, const f16* Kf,
                                                  const f16* KT, f16* cat) {
  __shared__ __align__(64) f16 psh[8][16 * 32];
  int w = (blockIdx.x << 3) + (threadIdx.x >> 5);
  if (w >= NH * (LQ / 16)) return;
  int h = w >> 7, t = w & 127;
  int lane = threadIdx.x & 31;
  f16* P = psh[threadIdx.x >> 5];

  const f16* Qb = Q + (long)(t * 16) * H + h * HD;
  v16h q0 = load_a(Qb, H);        // K-dim (HD) chunk 0..31
  v16h q1 = load_a(Qb + 32, H);   // chunk 32..63

  float m[8], l[8];
  v8f zero = {};
  v8f O[4];
  for (int j = 0; j < 8; ++j) { m[j] = -1e30f; l[j] = 0.f; }
  for (int d = 0; d < 4; ++d) O[d] = zero;

  for (int s0 = 0; s0 < SQ; s0 += 32) {
    // scores = Q * K^T / 8 for 16x32 key block (two 16x16 tiles)
    v8f sc0 = zero, sc1 = zero;
    const f16* Kb0 = Kf + (long)s0 * H + h * HD;
    const f16* Kb1 = Kf + (long)(s0 + 16) * H + h * HD;
    sc0 = wmma32(q0, load_b(Kb0, H), sc0);
    sc0 = wmma32(q1, load_b(Kb0 + 32, H), sc0);
    sc1 = wmma32(q0, load_b(Kb1, H), sc1);
    sc1 = wmma32(q1, load_b(Kb1 + 32, H), sc1);

    float nm[8], fac[8];
    for (int j = 0; j < 8; ++j) {
      float a = sc0[j] * 0.125f, b = sc1[j] * 0.125f;
      sc0[j] = a; sc1[j] = b;
      float bm = fmaxf(a, b);                       // row-max over 16 lanes
      bm = fmaxf(bm, __shfl_xor(bm, 1, 32));        // (xor 1,2,4,8 stays in
      bm = fmaxf(bm, __shfl_xor(bm, 2, 32));        //  each 16-lane half)
      bm = fmaxf(bm, __shfl_xor(bm, 4, 32));
      bm = fmaxf(bm, __shfl_xor(bm, 8, 32));
      nm[j] = fmaxf(m[j], bm);
      fac[j] = __expf(m[j] - nm[j]);
      m[j] = nm[j];
    }
    int col = lane & 15;
    for (int j = 0; j < 8; ++j) {
      float p0 = __expf(sc0[j] - nm[j]);
      float p1 = __expf(sc1[j] - nm[j]);
      int r = j + ((lane >> 4) << 3);
      P[r * 32 + col]      = (f16)p0;   // C-layout -> row-major via LDS
      P[r * 32 + 16 + col] = (f16)p1;
      float rs = p0 + p1;
      rs += __shfl_xor(rs, 1, 32);
      rs += __shfl_xor(rs, 2, 32);
      rs += __shfl_xor(rs, 4, 32);
      rs += __shfl_xor(rs, 8, 32);
      l[j] = l[j] * fac[j] + rs;
      for (int d = 0; d < 4; ++d) O[d][j] *= fac[j];
    }
    // same-wave LDS is in-order (DS hazard rules); re-read as A fragment
    v16h pf = load_a(P, 32);
    for (int d = 0; d < 4; ++d) {
      const f16* KTb = KT + (long)(h * HD + d * 16) * SQ + s0;
      O[d] = wmma32(pf, load_b(KTb, SQ), O[d]);
    }
  }

  int col = lane & 15;
  int r = h * 128 + t;
  for (int j = 0; j < 8; ++j) {
    float inv = 1.f / l[j];
    int i = j + ((lane >> 4) << 3);
    for (int d = 0; d < 4; ++d)
      cat[(long)r * (2 * H) + i * 64 + d * 16 + col] = (f16)(O[d][j] * inv);
  }
}

// ---------------- host ----------------------------------------------------

extern "C" void kernel_launch(void* const* d_in, const int* in_sizes, int n_in,
                              void* d_out, int out_size, void* d_ws,
                              size_t ws_size, hipStream_t stream) {
  const float* o    = (const float*)d_in[0];
  const float* f    = (const float*)d_in[1];
  const float* q0v  = (const float*)d_in[2];
  const float* q0g  = (const float*)d_in[3];
  const float* q0b  = (const float*)d_in[4];
  const float* q1v  = (const float*)d_in[5];
  const float* q1g  = (const float*)d_in[6];
  const float* q1b  = (const float*)d_in[7];
  const float* outv = (const float*)d_in[8];
  const float* outg = (const float*)d_in[9];
  const float* outb = (const float*)d_in[10];
  const float* Vv   = (const float*)d_in[11];
  const float* Vg   = (const float*)d_in[12];
  const float* Vb   = (const float*)d_in[13];
  const float* Cv   = (const float*)d_in[14];
  const float* Cg   = (const float*)d_in[15];
  const float* Cb   = (const float*)d_in[16];
  float* out = (float*)d_out;

  size_t off = 0;
  auto alloc = [&](size_t elems) -> f16* {
    f16* p = (f16*)((char*)d_ws + off);
    off += (elems * sizeof(f16) + 255) & ~(size_t)255;
    return p;
  };
  f16* w0  = alloc(3L * H * CINC);               // conv0 weights, 3 planes
  f16* w1  = alloc(3L * H * H);                  // conv1 weights
  f16* wo  = alloc((long)H * 2 * H);             // out proj
  f16* wV  = alloc((long)VOCABN * H);            // vocab head
  f16* wC  = alloc((long)LIMITN * H);            // copy head
  f16* xo  = alloc((long)(LQ + 2 * PAD) * CINC); // padded f16(o)
  f16* kf  = alloc((long)SQ * H);                // selu(f) (S,H)
  f16* kt  = alloc((long)H * SQ);                // selu(f)^T (H,S)
  f16* x1  = alloc((long)(LQ + 2 * PAD) * H);    // padded conv0 out
  f16* q   = alloc((long)LQ * H);                // conv1 out (queries)
  f16* cat = alloc((long)LQ * 2 * H);            // [mix | q2] permuted rows
  f16* ao  = alloc((long)LQ * H);                // selu(att_out)

  dim3 blk(256);
  // fold weight norm, convert to f16
  wn_rows<<<VOCABN, blk, 0, stream>>>(Vv, Vg, wV, H);
  wn_rows<<<LIMITN, blk, 0, stream>>>(Cv, Cg, wC, H);
  wn_rows<<<H, blk, 0, stream>>>(outv, outg, wo, 2 * H);
  wn_conv<<<H, blk, 0, stream>>>(q0v, q0g, w0, CINC, H);
  wn_conv<<<H, blk, 0, stream>>>(q1v, q1g, w1, H, H);
  // activations
  long no = (long)(LQ + 2 * PAD) * CINC;
  prep_o<<<(int)((no + 255) / 256), blk, 0, stream>>>(o, xo);
  long nk = (long)SQ * H;
  prep_kv<<<(int)((nk + 255) / 256), blk, 0, stream>>>(f, kf, kt);
  zero_h<<<(PAD * H + 255) / 256, blk, 0, stream>>>(x1, PAD * H);
  zero_h<<<(PAD * H + 255) / 256, blk, 0, stream>>>(x1 + (long)(LQ + PAD) * H,
                                                    PAD * H);
  // conv stack (3-shift WMMA GEMMs) + selu; 32x64 tiles per wave
  int tc = (LQ / 32) * (H / 64);
  conv_gemm<<<(tc + 7) / 8, blk, 0, stream>>>(xo, w0, q0b,
                                              x1 + (long)PAD * H, CINC, H / 64);
  conv_gemm<<<(tc + 7) / 8, blk, 0, stream>>>(x1, w1, q1b, q, H, H / 64);
  // pack q2 half of cat, then attention writes mix half
  long nq = (long)LQ * H;
  q2pack<<<(int)((nq + 255) / 256), blk, 0, stream>>>(q, cat);
  attn_flash<<<(NH * (LQ / 16)) / 8, blk, 0, stream>>>(q, kf, kt, cat);
  // output projection + selu
  gemm_ws<<<(tc + 7) / 8, blk, 0, stream>>>(cat, 2 * H, wo, 2 * H, outb, 2 * H,
                                            LQ / 32, H / 64, ao, nullptr, H, 1);
  // logits: (L x 32000) and (L x 512) into fused (L x 32512) f32 output
  int t2 = (LQ / 32) * (VOCABN / 64);
  gemm_ws<<<(t2 + 7) / 8, blk, 0, stream>>>(ao, H, wV, H, Vb, H, LQ / 32,
                                            VOCABN / 64, nullptr, out,
                                            VOCABN + LIMITN, 0);
  int t3 = (LQ / 32) * (LIMITN / 64);
  gemm_ws<<<(t3 + 7) / 8, blk, 0, stream>>>(ao, H, wC, H, Cb, H, LQ / 32,
                                            LIMITN / 64, nullptr, out + VOCABN,
                                            VOCABN + LIMITN, 0);
}
